// LSTM_74749610819710
// MI455X (gfx1250) — matrix operand order repaired and checked by reading, not compiled, not run on hardware
//
#include <hip/hip_runtime.h>
#include <hip/hip_bf16.h>

// ---------------- problem constants ----------------
#define NB    64          // batch
#define TT    1024        // timesteps
#define DD    512         // input dim
#define HH    512         // hidden dim
#define FF    2048        // 4*H
#define NT    (NB*TT)     // 65536 rows of the input-projection GEMM
#define LDSW  520         // padded LDS row stride (elements) for h buffer

// ---------------- types ----------------
typedef __bf16        v16bf  __attribute__((ext_vector_type(16)));
typedef float         v8f    __attribute__((ext_vector_type(8)));
typedef unsigned int  uint4v __attribute__((ext_vector_type(4)));

union FragBF {
    v16bf  v;
    uint4v u4[2];
};

__device__ __forceinline__ unsigned short f32_to_bf16(float f) {
    unsigned int u = __float_as_uint(f);
    u += 0x7FFFu + ((u >> 16) & 1u);      // round-to-nearest-even
    return (unsigned short)(u >> 16);
}

__device__ __forceinline__ float sigm(float x) {
    return 1.0f / (1.0f + __expf(-x));
}

// ---------------- phase 0a: x (f32) -> bf16 ----------------
__global__ void k_cvt_x(const float* __restrict__ x, unsigned short* __restrict__ xb, int n) {
    int i = blockIdx.x * blockDim.x + threadIdx.x;
    if (i < n) xb[i] = f32_to_bf16(x[i]);
}

// ---------------- phase 0b: W (rows x cols, f32) -> WT (cols x rows, bf16) ----------------
__global__ void k_transpose_bf16(const float* __restrict__ W, unsigned short* __restrict__ WT,
                                 int rows, int cols) {
    int idx = blockIdx.x * blockDim.x + threadIdx.x;
    if (idx < rows * cols) {
        int r = idx / cols;
        int c = idx - r * cols;
        WT[(size_t)c * rows + r] = f32_to_bf16(W[(size_t)r * cols + c]);
    }
}

// ---------------- phase 1: xW = x @ Wx + b  (bf16 WMMA, f32 accum) ----------------
// grid = (NT/16, 2), block = 256 (8 waves). Wave computes a 16x128 strip of C.
__global__ __launch_bounds__(256)
void k_xw_gemm(const unsigned short* __restrict__ xb,   // (NT, D)  bf16 row-major
               const unsigned short* __restrict__ WxT,  // (F, D)   bf16 row-major (transposed Wx)
               const float*          __restrict__ bias, // (F)
               float*                __restrict__ xW)   // (NT, F)  f32
{
    const int lane = threadIdx.x & 31;
    const int wave = threadIdx.x >> 5;
    const int half = lane >> 4;      // 0|1
    const int ln   = lane & 15;      // A-row m / B-col n / C-col n
    const int rowBase = blockIdx.x * 16;
    const int colBase = (blockIdx.y * 8 + wave) * 128;

    v8f acc[8];
#pragma unroll
    for (int j = 0; j < 8; ++j)
#pragma unroll
        for (int v = 0; v < 8; ++v) acc[j][v] = 0.0f;

    const unsigned short* aRow = xb + (size_t)(rowBase + ln) * DD;

    for (int kk = 0; kk < DD / 32; ++kk) {
        // A fragment (16x32 bf16): lane holds row ln, K = 32kk+16*half+{0..15}
        FragBF a;
        const unsigned short* ap = aRow + kk * 32 + half * 16;
        a.u4[0] = *(const uint4v*)(ap);
        a.u4[1] = *(const uint4v*)(ap + 8);
#pragma unroll
        for (int j = 0; j < 8; ++j) {
            // B fragment (32x16 bf16): lane holds col (tile base + ln), K = 32kk+16*half+{0..15}
            const unsigned short* bp =
                WxT + (size_t)(colBase + j * 16 + ln) * DD + kk * 32 + half * 16;
            FragBF b;
            b.u4[0] = *(const uint4v*)(bp);
            b.u4[1] = *(const uint4v*)(bp + 8);
            acc[j] = __builtin_amdgcn_wmma_f32_16x16x32_bf16(
                false, a.v, false, b.v, (short)0, acc[j], false, false);
        }
    }

#pragma unroll
    for (int j = 0; j < 8; ++j) {
        const int col = colBase + j * 16 + ln;        // C col: N = lane&15
        const float bv = bias[col];
#pragma unroll
        for (int v = 0; v < 8; ++v) {
            const int mr = v + 8 * half;              // C row: M = v + 8*(lane>>4)
            xW[(size_t)(rowBase + mr) * FF + col] = acc[j][v] + bv;
        }
    }
}

// ---------------- phase 2: sequential LSTM recurrence ----------------
// grid = N/16 = 4 workgroups; block = 512 threads = 16 waves.
// Wave w owns hidden units [32w, 32w+32): gate columns {k, H+k, 2H+k, 3H+k}.
// c-state lives in VGPRs; h-state double-buffered in LDS (bf16).
__global__ __launch_bounds__(512)
void k_lstm_rec(const float*          __restrict__ xW,   // (N*T, F) f32, bias included
                const unsigned short* __restrict__ WhT,  // (F, H) bf16 (transposed Wh)
                const float*          __restrict__ h0,   // (N, H)
                float*                __restrict__ out)  // (N, T, H)
{
    __shared__ alignas(16) unsigned short hbuf[2][16][LDSW];

    const int lane = threadIdx.x & 31;
    const int wave = threadIdx.x >> 5;   // 0..15
    const int half = lane >> 4;
    const int ln   = lane & 15;
    const int n0   = blockIdx.x * 16;    // this block's batch rows
    const int ku0  = wave * 32;          // this wave's hidden-unit base

    // load h0 -> hbuf[0] as bf16
    for (int i = threadIdx.x; i < 16 * HH; i += blockDim.x) {
        int mm = i >> 9;                 // /HH
        int k  = i & (HH - 1);
        hbuf[0][mm][k] = f32_to_bf16(h0[(size_t)(n0 + mm) * HH + k]);
    }
    __syncthreads();

    float cst[2][8];
#pragma unroll
    for (int j = 0; j < 2; ++j)
#pragma unroll
        for (int v = 0; v < 8; ++v) cst[j][v] = 0.0f;

    for (int t = 0; t < TT; ++t) {
        const int buf = t & 1;

        // accumulators start at the precomputed input projections
        v8f acc[8];
#pragma unroll
        for (int j = 0; j < 8; ++j) {
            const int col = ku0 + (j & 1) * 16 + (j >> 1) * HH + ln;
#pragma unroll
            for (int v = 0; v < 8; ++v) {
                const int mr = v + 8 * half;
                acc[j][v] = xW[((size_t)(n0 + mr) * TT + t) * FF + col];
            }
        }
        // warm next timestep's xW rows in GL2 (global_prefetch_b8)
        if (t + 1 < TT) {
#pragma unroll
            for (int j = 0; j < 8; ++j) {
                const int col = ku0 + (j & 1) * 16 + (j >> 1) * HH + ln;
                __builtin_prefetch(&xW[((size_t)(n0 + 8 * half) * TT + (t + 1)) * FF + col], 0, 0);
            }
        }

        // acc += h_t @ Wh : A = h (16xH) from LDS, B = WhT rows from L2
        for (int kk = 0; kk < HH / 32; ++kk) {
            FragBF a;
            const unsigned short* ap = &hbuf[buf][ln][kk * 32 + half * 16];
            a.u4[0] = *(const uint4v*)(ap);
            a.u4[1] = *(const uint4v*)(ap + 8);
#pragma unroll
            for (int j = 0; j < 8; ++j) {
                const int col = ku0 + (j & 1) * 16 + (j >> 1) * HH + ln;
                const unsigned short* bp =
                    WhT + (size_t)col * HH + kk * 32 + half * 16;
                FragBF b;
                b.u4[0] = *(const uint4v*)(bp);
                b.u4[1] = *(const uint4v*)(bp + 8);
                acc[j] = __builtin_amdgcn_wmma_f32_16x16x32_bf16(
                    false, a.v, false, b.v, (short)0, acc[j], false, false);
            }
        }

        // gates + state update: i=acc[j], f=acc[j+2], o=acc[j+4], g=acc[j+6]
        const int nbuf = buf ^ 1;
#pragma unroll
        for (int j = 0; j < 2; ++j) {
            const int k = ku0 + j * 16 + ln;
#pragma unroll
            for (int v = 0; v < 8; ++v) {
                const int mr = v + 8 * half;
                const float iv = sigm(acc[j][v]);
                const float fv = sigm(acc[j + 2][v]);
                const float ov = sigm(acc[j + 4][v]);
                const float gv = tanhf(acc[j + 6][v]);
                const float cn = fv * cst[j][v] + iv * gv;
                cst[j][v] = cn;
                const float hv = ov * tanhf(cn);
                hbuf[nbuf][mr][k] = f32_to_bf16(hv);
                out[((size_t)(n0 + mr) * TT + t) * HH + k] = hv;
            }
        }
        __syncthreads();
    }
}

// ---------------- launch ----------------
extern "C" void kernel_launch(void* const* d_in, const int* in_sizes, int n_in,
                              void* d_out, int out_size, void* d_ws, size_t ws_size,
                              hipStream_t stream) {
    const float* x  = (const float*)d_in[0];   // (N, T, D)
    const float* h0 = (const float*)d_in[1];   // (N, H)
    const float* Wx = (const float*)d_in[2];   // (D, 4H)
    const float* Wh = (const float*)d_in[3];   // (H, 4H)
    const float* b  = (const float*)d_in[4];   // (4H)
    float* out = (float*)d_out;                // (N, T, H)

    // workspace layout (bytes): xb | WxT | WhT | xW   (~580 MB total)
    char* ws = (char*)d_ws;
    unsigned short* xb  = (unsigned short*)(ws);                       // NT*D bf16  = 64 MiB
    unsigned short* WxT = (unsigned short*)(ws + (size_t)NT * DD * 2); // F*D  bf16  = 2 MiB
    unsigned short* WhT = (unsigned short*)(ws + (size_t)NT * DD * 2 + (size_t)FF * DD * 2);
    float* xW = (float*)(ws + (size_t)NT * DD * 2 + 2 * (size_t)FF * DD * 2); // NT*F f32 = 512 MiB

    // phase 0: conversions / transposes
    {
        int n = NT * DD;
        k_cvt_x<<<(n + 255) / 256, 256, 0, stream>>>(x, xb, n);
        int nw = DD * FF;
        k_transpose_bf16<<<(nw + 255) / 256, 256, 0, stream>>>(Wx, WxT, DD, FF);
        k_transpose_bf16<<<(nw + 255) / 256, 256, 0, stream>>>(Wh, WhT, HH, FF);
    }

    // phase 1: xW = x @ Wx + b
    {
        dim3 grid(NT / 16, 2);
        k_xw_gemm<<<grid, 256, 0, stream>>>(xb, WxT, b, xW);
    }

    // phase 2: recurrence (4 workgroups, c-state in VGPRs, h-state in LDS)
    {
        k_lstm_rec<<<NB / 16, 512, 0, stream>>>(xW, WhT, h0, out);
    }
}
